// SWAttention_38886633898107
// MI455X (gfx1250) — matrix operand order repaired and checked by reading, not compiled, not run on hardware
//
#include <hip/hip_runtime.h>

// Problem constants (from reference): B=2, T=2048, Dm=2048, H=32, D=64, WIN=512
#define HB    32
#define DH    64
#define WINSZ 512
#define TT    2048
#define BB    2
#define BT    4096   // B*T
#define DM    2048   // model dim
#define NB    4      // T/WIN

typedef __attribute__((ext_vector_type(16))) __bf16 v16bf;
typedef __attribute__((ext_vector_type(8)))  __bf16 v8bf;
typedef __attribute__((ext_vector_type(8)))  float  v8f;

__device__ __forceinline__ __bf16 f2bf(float f) {
  unsigned u = __builtin_bit_cast(unsigned, f);
  u += 0x7FFFu + ((u >> 16) & 1u);           // round-to-nearest-even
  unsigned short s = (unsigned short)(u >> 16);
  return __builtin_bit_cast(__bf16, s);
}

// A-operand 16x32 bf16 tile from row-major [M][K] memory (ISA A layout:
// lanes 0-15 hold K 0..7 / 16..23, lanes 16-31 hold K 8..15 / 24..31)
__device__ __forceinline__ v16bf load_A16x32(const __bf16* base, size_t ld, int lane) {
  int r  = lane & 15;
  int kb = (lane & 16) ? 8 : 0;
  const __bf16* p = base + (size_t)r * ld + kb;
  union { v16bf v; v8bf h[2]; } u;
  u.h[0] = *(const v8bf*)(p);
  u.h[1] = *(const v8bf*)(p + 16);
  return u.v;
}

// B-operand 32x16 bf16 tile from Bt = B^T stored row-major [N][K] (ISA B layout:
// lane n = column N, lanes 0-15 hold K 0..7 / 16..23, lanes 16-31 hold K 8..15 / 24..31)
__device__ __forceinline__ v16bf load_B32x16(const __bf16* baseT, size_t ld, int lane) {
  int n  = lane & 15;
  int kb = (lane & 16) ? 8 : 0;
  const __bf16* p = baseT + (size_t)n * ld + kb;
  union { v16bf v; v8bf h[2]; } u;
  u.h[0] = *(const v8bf*)(p);
  u.h[1] = *(const v8bf*)(p + 16);
  return u.v;
}

__device__ __forceinline__ v8f wmma_bf16(v16bf a, v16bf b, v8f c) {
  return __builtin_amdgcn_wmma_f32_16x16x32_bf16(false, a, false, b, (short)0, c,
                                                 false, false);
}

// CDNA5 async global -> LDS copy, 16B per lane, tracked by ASYNCcnt.
__device__ __forceinline__ void async_load_b128(unsigned lds_off,
                                                unsigned long long gaddr) {
  asm volatile("global_load_async_to_lds_b128 %0, %1, off"
               :: "v"(lds_off), "v"(gaddr) : "memory");
}
__device__ __forceinline__ void wait_asynccnt0() {
  asm volatile("s_wait_asynccnt 0x0" ::: "memory");
}

// ---------------------------------------------------------------- cast x -> bf16
__global__ void cast_f32_bf16(const float* __restrict__ in, __bf16* __restrict__ out,
                              int n) {
  int i = blockIdx.x * blockDim.x + threadIdx.x;
  if (i < n) out[i] = f2bf(in[i]);
}

// -------------------------------------------- weight transpose+cast: Wt[n][k]=W[k][n]
__global__ void transpose_cast_w(const float* __restrict__ W, __bf16* __restrict__ Wt) {
  __shared__ float tile[32][33];
  int x  = blockIdx.x * 32 + threadIdx.x;
  int y0 = blockIdx.y * 32;
  for (int i = threadIdx.y; i < 32; i += 8)
    tile[i][threadIdx.x] = W[(size_t)(y0 + i) * DM + x];
  __syncthreads();
  int xo  = blockIdx.y * 32 + threadIdx.x;
  int yo0 = blockIdx.x * 32;
  for (int i = threadIdx.y; i < 32; i += 8)
    Wt[(size_t)(yo0 + i) * DM + xo] = f2bf(tile[threadIdx.x][i]);
}

// ------------------------------------------------- DoG bias table [H][WIN] (fp32)
__device__ __forceinline__ float softplusf(float x) {
  return (x > 20.f) ? x : log1pf(__expf(x));
}

__global__ void dog_bias_kernel(const float* __restrict__ dog, float* __restrict__ tab) {
  int i = blockIdx.x * blockDim.x + threadIdx.x;
  if (i >= HB * WINSZ) return;
  int h = i >> 9, dist = i & (WINSZ - 1);
  const float* p = dog + h * 7;
  float a1 = p[0], s1 = softplusf(p[1]), m1 = p[2];
  float a2 = p[3], s2 = softplusf(p[4]), m2 = p[5];
  float c = p[6];
  float d  = (float)dist;
  float g1 = a1 * __expf(-(d - m1) * (d - m1) / (2.f * s1 * s1));
  float g2 = a2 * __expf(-(d - m2) * (d - m2) / (2.f * s2 * s2));
  tab[i] = g1 - g2 + c;
}

// ----------------------------------------------------------------- WMMA GEMM
// C[M=4096][N=2048] = A[M][K=2048](bf16) x B, given Bt[N][K](bf16).
// The 64x32 B panel is shared by all 8 waves of the block -> staged through LDS
// with double-buffered async global->LDS copies (ASYNCcnt).
// MODE 0: bf16 row-major out; MODE 1: fp32 row-major out;
// MODE 2: bf16 out in [B][H][D][T] transposed layout (for V).
template <int MODE>
__global__ void __launch_bounds__(256) gemm_bf16_wmma(const __bf16* __restrict__ A,
                                                      const __bf16* __restrict__ Bt,
                                                      void* __restrict__ Cout) {
  __shared__ __align__(16) __bf16 bstage[2][64 * 32];
  const int tid  = threadIdx.x;
  const int lane = tid & 31;
  const int w    = tid >> 5;
  const int mt   = blockIdx.x & 31;   // M/128 = 32
  const int nt   = blockIdx.x >> 5;   // N/64  = 32
  const int row0 = mt * 128 + w * 16;
  const int n0   = nt * 64;

  // async staging: thread -> (row sn of 64, 16B part sp of 4)
  const int sn = tid >> 2;
  const int sp = tid & 3;
  const __bf16* gB = Bt + (size_t)(n0 + sn) * DM + sp * 8;
  const unsigned ldsB0 = (unsigned)(uintptr_t)&bstage[0][sn * 32 + sp * 8];
  const unsigned ldsB1 = (unsigned)(uintptr_t)&bstage[1][sn * 32 + sp * 8];

  async_load_b128(ldsB0, (unsigned long long)(uintptr_t)gB);
  wait_asynccnt0();
  __syncthreads();

  v8f acc[4] = {};
  for (int kk = 0; kk < DM; kk += 32) {
    const int cur = (kk >> 5) & 1;
    if (kk + 32 < DM)
      async_load_b128(cur ? ldsB0 : ldsB1,
                      (unsigned long long)(uintptr_t)(gB + kk + 32));

    v16bf a = load_A16x32(A + (size_t)row0 * DM + kk, DM, lane);
    const __bf16* bs = &bstage[cur][0];
#pragma unroll
    for (int c = 0; c < 4; ++c) {
      v16bf b = load_B32x16(bs + c * 16 * 32, 32, lane);
      acc[c] = wmma_bf16(a, b, acc[c]);
    }
    wait_asynccnt0();
    __syncthreads();
  }

  const int mrow = row0 + ((lane >> 4) << 3);
  const int ncol = n0 + (lane & 15);
#pragma unroll
  for (int c = 0; c < 4; ++c) {
#pragma unroll
    for (int r = 0; r < 8; ++r) {
      int rr = mrow + r, cc = ncol + c * 16;
      if (MODE == 1) {
        ((float*)Cout)[(size_t)rr * DM + cc] = acc[c][r];
      } else if (MODE == 0) {
        ((__bf16*)Cout)[(size_t)rr * DM + cc] = f2bf(acc[c][r]);
      } else {  // V transposed: vt[((b*H+h)*D+d)*T + t]
        int b = rr >> 11, t = rr & (TT - 1);
        int h = cc >> 6, d = cc & (DH - 1);
        ((__bf16*)Cout)[((size_t)(b * HB + h) * DH + d) * TT + t] = f2bf(acc[c][r]);
      }
    }
  }
}

// --------------------------------------------------- sliding-window attention
// One wave per (b, block, head, 16-row q strip). Flash-style online softmax.
// Row sums are produced on the matrix pipe via an extra L = P @ ones WMMA
// accumulator (rescales with alpha automatically), so only the row-max needs
// cross-lane reductions.
__global__ void __launch_bounds__(128) sw_attn_wmma(const __bf16* __restrict__ q,
                                                    const __bf16* __restrict__ kmat,
                                                    const __bf16* __restrict__ vt,
                                                    const float* __restrict__ bias_tab,
                                                    const float* __restrict__ sinks,
                                                    __bf16* __restrict__ attn) {
  __shared__ __align__(16) __bf16 pstage[4][16 * 32];
  const int lane = threadIdx.x & 31;
  const int w    = threadIdx.x >> 5;
  const int wid  = blockIdx.x * 4 + w;
  const int qt   = wid & 31;          // q strip within block (WIN/16 = 32)
  const int h    = (wid >> 5) & 31;
  const int nblk = (wid >> 10) & 3;
  const int b    = wid >> 12;

  const float scale = 0.125f;  // D^-0.5
  const int   qlo   = qt * 16;
  const size_t t0   = (size_t)b * TT + nblk * WINSZ + qlo;

  const __bf16* qbase = q + t0 * DM + h * DH;
  v16bf Aq0 = load_A16x32(qbase, DM, lane);       // d 0..31
  v16bf Aq1 = load_A16x32(qbase + 32, DM, lane);  // d 32..63

  v16bf ones;
#pragma unroll
  for (int i = 0; i < 16; ++i) ones[i] = f2bf(1.0f);

  v8f O[4] = {};
  v8f Lacc = {};
  float mrow[8];
#pragma unroll
  for (int r = 0; r < 8; ++r) mrow[r] = -3.0e38f;

  const int rowadd = (lane >> 4) << 3;
  const float* btab = bias_tab + h * WINSZ;
  const __bf16* vth = vt + (size_t)(b * HB + h) * DH * TT;
  __bf16* ps = pstage[w];

  int jstart = (qlo >> 5) << 5;
  if (nblk == 0 && jstart < WINSZ) jstart = WINSZ;     // no previous block
  const int jend = ((qlo + WINSZ + 15) >> 5) << 5;     // last 32-key tile start

  for (int j = jstart; j <= jend; j += 32) {
    const int tk0 = nblk * WINSZ + j - WINSZ;          // timeline index of key j
    const __bf16* kb0 = kmat + ((size_t)b * TT + tk0) * DM + h * DH;

    v16bf B00 = load_B32x16(kb0, DM, lane);                 // keys j..j+15, d 0..31
    v16bf B01 = load_B32x16(kb0 + 32, DM, lane);            // keys j..j+15, d 32..63
    v16bf B10 = load_B32x16(kb0 + (size_t)16 * DM, DM, lane);
    v16bf B11 = load_B32x16(kb0 + (size_t)16 * DM + 32, DM, lane);

    v8f s0 = {}, s1 = {};
    s0 = wmma_bf16(Aq0, B00, s0);
    s0 = wmma_bf16(Aq1, B01, s0);
    s1 = wmma_bf16(Aq0, B10, s1);
    s1 = wmma_bf16(Aq1, B11, s1);

    // scale + DoG bias + window mask. Bias gathers are made unconditional by
    // clamping the distance into [0, WIN) (always a legal address), then the
    // mask is applied with a branch-free select -> no exec-mask branches.
    float sa0[8], sa1[8];
    const int kc0 = j + (lane & 15);
#pragma unroll
    for (int r = 0; r < 8; ++r) {
      int qp = qlo + rowadd + r;
      int d0 = qp - kc0 + WINSZ;   // relative distance, valid in [0, WIN)
      int d1 = d0 - 16;
      int c0 = min(max(d0, 0), WINSZ - 1);
      int c1 = min(max(d1, 0), WINSZ - 1);
      float bv0 = btab[c0];
      float bv1 = btab[c1];
      sa0[r] = ((unsigned)d0 < (unsigned)WINSZ) ? (s0[r] * scale + bv0) : -1.0e30f;
      sa1[r] = ((unsigned)d1 < (unsigned)WINSZ) ? (s1[r] * scale + bv1) : -1.0e30f;
    }

    // online softmax: per-row max across the 16-lane N groups
    float p0[8], p1[8], al[8];
    int changed = 0;
#pragma unroll
    for (int r = 0; r < 8; ++r) {
      float vmax = fmaxf(sa0[r], sa1[r]);
      vmax = fmaxf(vmax, __shfl_xor(vmax, 1));
      vmax = fmaxf(vmax, __shfl_xor(vmax, 2));
      vmax = fmaxf(vmax, __shfl_xor(vmax, 4));
      vmax = fmaxf(vmax, __shfl_xor(vmax, 8));
      float mold = mrow[r];
      float mnew = fmaxf(mold, vmax);
      al[r] = __expf(mold - mnew);
      changed |= (mnew > mold) ? 1 : 0;
      mrow[r] = mnew;
      p0[r] = __expf(sa0[r] - mnew);
      p1[r] = __expf(sa1[r] - mnew);
    }
    if (__any(changed)) {
#pragma unroll
      for (int r = 0; r < 8; ++r) {
        float a = al[r];
#pragma unroll
        for (int c = 0; c < 4; ++c) O[c][r] = O[c][r] * a;
        Lacc[r] = Lacc[r] * a;
      }
    }

    // re-layout P (C-layout -> A-layout) through per-wave LDS tile
    {
      int col = lane & 15;
#pragma unroll
      for (int r = 0; r < 8; ++r) {
        int row = rowadd + r;
        ps[row * 32 + col]      = f2bf(p0[r]);
        ps[row * 32 + 16 + col] = f2bf(p1[r]);
      }
    }
    asm volatile("s_wait_dscnt 0x0" ::: "memory");
    v16bf Ap = load_A16x32(ps, 32, lane);
    asm volatile("" ::: "memory");  // keep next iter's LDS stores after this load

    // O += P @ V ; L += P @ ones (row sums on the matrix pipe)
#pragma unroll
    for (int c = 0; c < 4; ++c) {
      v16bf Bv = load_B32x16(vth + (size_t)(c * 16) * TT + tk0, TT, lane);
      O[c] = wmma_bf16(Ap, Bv, O[c]);
    }
    Lacc = wmma_bf16(Ap, ones, Lacc);
  }

  // attention sink folded into normalization: m = max(mrow, sink)
  float sk = sinks[h];
  float rs[8];
#pragma unroll
  for (int r = 0; r < 8; ++r) {
    float mf    = fmaxf(mrow[r], sk);
    float a2    = __expf(mrow[r] - mf);
    float denom = Lacc[r] * a2 + __expf(sk - mf);
    rs[r] = a2 / denom;
  }
  __bf16* outp = attn + t0 * DM + h * DH;
  int col = lane & 15;
#pragma unroll
  for (int c = 0; c < 4; ++c)
#pragma unroll
    for (int r = 0; r < 8; ++r)
      outp[(size_t)(rowadd + r) * DM + c * 16 + col] = f2bf(O[c][r] * rs[r]);
}

// ------------------------------------------------------------------- launcher
extern "C" void kernel_launch(void* const* d_in, const int* in_sizes, int n_in,
                              void* d_out, int out_size, void* d_ws, size_t ws_size,
                              hipStream_t stream) {
  (void)in_sizes; (void)n_in; (void)out_size; (void)ws_size;
  const float* x     = (const float*)d_in[0];
  const float* Wq    = (const float*)d_in[1];
  const float* Wk    = (const float*)d_in[2];
  const float* Wv    = (const float*)d_in[3];
  const float* Wo    = (const float*)d_in[4];
  const float* dog   = (const float*)d_in[5];
  const float* sinks = (const float*)d_in[6];

  char* ws = (char*)d_ws;
  size_t off = 0;
  auto alloc = [&](size_t bytes) -> void* {
    void* p = ws + off;
    off = (off + bytes + 255) & ~(size_t)255;
    return p;
  };
  __bf16* xb   = (__bf16*)alloc((size_t)BT * DM * 2);
  __bf16* Wqt  = (__bf16*)alloc((size_t)DM * DM * 2);
  __bf16* Wkt  = (__bf16*)alloc((size_t)DM * DM * 2);
  __bf16* Wvt  = (__bf16*)alloc((size_t)DM * DM * 2);
  __bf16* Wot  = (__bf16*)alloc((size_t)DM * DM * 2);
  __bf16* qb   = (__bf16*)alloc((size_t)BT * DM * 2);
  __bf16* kb   = (__bf16*)alloc((size_t)BT * DM * 2);
  __bf16* vtb  = (__bf16*)alloc((size_t)BT * DM * 2);   // [B][H][D][T]
  __bf16* attn = (__bf16*)alloc((size_t)BT * DM * 2);
  float*  btab = (float*)alloc((size_t)HB * WINSZ * 4);

  const int n = BT * DM;
  cast_f32_bf16<<<n / 256, 256, 0, stream>>>(x, xb, n);

  dim3 tg(64, 64), tb(32, 8);
  transpose_cast_w<<<tg, tb, 0, stream>>>(Wq, Wqt);
  transpose_cast_w<<<tg, tb, 0, stream>>>(Wk, Wkt);
  transpose_cast_w<<<tg, tb, 0, stream>>>(Wv, Wvt);
  transpose_cast_w<<<tg, tb, 0, stream>>>(Wo, Wot);

  dog_bias_kernel<<<(HB * WINSZ) / 256, 256, 0, stream>>>(dog, btab);

  gemm_bf16_wmma<0><<<1024, 256, 0, stream>>>(xb, Wqt, qb);
  gemm_bf16_wmma<0><<<1024, 256, 0, stream>>>(xb, Wkt, kb);
  gemm_bf16_wmma<2><<<1024, 256, 0, stream>>>(xb, Wvt, vtb);

  // 2 batches * 4 blocks * 32 heads * 32 q-strips = 8192 waves, 4 waves/block
  sw_attn_wmma<<<2048, 128, 0, stream>>>(qb, kb, vtb, btab, sinks, attn);

  gemm_bf16_wmma<1><<<1024, 256, 0, stream>>>(attn, Wot, (float*)d_out);
}
